// GAT_18013092839755
// MI455X (gfx1250) — matrix operand order repaired and checked by reading, not compiled, not run on hardware
//
#include <hip/hip_runtime.h>

#define IN_CH 768
#define HID   256
#define HEADS 4
#define HDIM  64

typedef __attribute__((ext_vector_type(16))) __bf16 v16bf;
typedef __attribute__((ext_vector_type(8)))  __bf16 v8bf;
typedef __attribute__((ext_vector_type(8)))  float  v8f;

// ---------------------------------------------------------------- utilities
__device__ __forceinline__ unsigned fenc(float f) {
    unsigned u = __float_as_uint(f);
    return (u & 0x80000000u) ? ~u : (u | 0x80000000u);   // monotonic float->uint
}
__device__ __forceinline__ float fdec(unsigned u) {
    unsigned b = (u & 0x80000000u) ? (u & 0x7FFFFFFFu) : ~u;
    return __uint_as_float(b);
}

// ---------------------------------------------------------------- f32 -> bf16
__global__ __launch_bounds__(256) void gat_cvt_bf16(const float* __restrict__ in,
                                                    __bf16* __restrict__ out,
                                                    long n) {
    long i = (long)blockIdx.x * 256 + threadIdx.x;
    if (i < n) out[i] = (__bf16)in[i];
}

// ------------------------------------------- f32 [K x 256] -> bf16^T [256 x K]
__global__ __launch_bounds__(256) void gat_cvt_bf16_t(const float* __restrict__ in,
                                                      __bf16* __restrict__ out,
                                                      int K) {
    long i = (long)blockIdx.x * 256 + threadIdx.x;
    if (i < (long)K * HID) {
        int k = (int)(i >> 8);          // i / 256
        int n = (int)(i & 255);         // i % 256
        out[(size_t)n * K + k] = (__bf16)in[i];
    }
}

// ---------------------------------------------------------------- WMMA GEMM
// C[M x 256] = act(A[Mp x K](bf16,row-major, Mp padded to 32) @ Bt[256 x K] + bias)
// Block tile 32 x 256, 256 threads = 8 waves: wave w -> rows rg=w>>2 (16 rows),
// cols cg=w&3 (64 cols = 4 WMMA tiles). No LDS, no load guards (A buffer rows
// are padded; garbage rows only produce garbage output rows that are never
// stored). All 5 fragment loads issued before the 4 WMMAs each K-step.
__global__ __launch_bounds__(256) void gat_gemm_bf16(const __bf16* __restrict__ A,
                                                     const __bf16* __restrict__ Bt,
                                                     const float*  __restrict__ bias,
                                                     float*  __restrict__ Cf,
                                                     __bf16* __restrict__ Cb,
                                                     int M, int K, float slope) {
    const int t    = threadIdx.x;
    const int w    = t >> 5;
    const int lane = t & 31;
    const int lid  = lane & 15;
    const int hi   = lane >> 4;
    const int rg   = w >> 2;             // row group 0..1
    const int cg   = w & 3;              // col group 0..3
    const int rowBase = blockIdx.x * 32 + rg * 16;
    const __bf16* ap = A + (size_t)(rowBase + lid) * K + hi * 16; // chunks hi*16, hi*16+8

    const __bf16* bp[4];
#pragma unroll
    for (int c = 0; c < 4; ++c)
        bp[c] = Bt + (size_t)(cg * 64 + c * 16 + lid) * K + hi * 16;

    v8f acc[4];
#pragma unroll
    for (int c = 0; c < 4; ++c)
#pragma unroll
        for (int r = 0; r < 8; ++r) acc[c][r] = 0.0f;

#pragma unroll 2
    for (int k0 = 0; k0 < K; k0 += 32) {
        v8bf a0 = *(const v8bf*)(ap + k0);
        v8bf a1 = *(const v8bf*)(ap + k0 + 8);
        v16bf bfrag[4];
#pragma unroll
        for (int c = 0; c < 4; ++c)
            bfrag[c] = *(const v16bf*)(bp[c] + k0);   // K = hi*16 + j (contiguous)
        v16bf a = __builtin_shufflevector(a0, a1, 0, 1, 2, 3, 4, 5, 6, 7,
                                                  8, 9, 10, 11, 12, 13, 14, 15);
#pragma unroll
        for (int c = 0; c < 4; ++c)
            acc[c] = __builtin_amdgcn_wmma_f32_16x16x32_bf16(
                false, a, false, bfrag[c], (short)0, acc[c], false, false);
    }

    // store: vgpr r, lane -> row rowBase + hi*8 + r, col = cg*64 + c*16 + lid
#pragma unroll
    for (int c = 0; c < 4; ++c) {
        int col  = cg * 64 + c * 16 + lid;
        float bv = bias ? bias[col] : 0.0f;
#pragma unroll
        for (int r = 0; r < 8; ++r) {
            int row = rowBase + hi * 8 + r;
            if (row < M) {
                float v = acc[c][r] + bv;
                v = (v < 0.0f) ? v * slope : v;
                if (Cf) Cf[(size_t)row * HID + col] = v;
                if (Cb) Cb[(size_t)row * HID + col] = (__bf16)v;
            }
        }
    }
}

// ------------------------------------------- per-node attention coefficients
// block = 128 threads = 4 waves, one wave per head, one block per node
__global__ __launch_bounds__(128) void gat_att_scores(const float* __restrict__ h,
                                                      const float* __restrict__ att_s,
                                                      const float* __restrict__ att_d,
                                                      float* __restrict__ asrc,
                                                      float* __restrict__ adst,
                                                      int N) {
    int n = blockIdx.x;
    if (n >= N) return;
    int hd   = threadIdx.x >> 5;
    int lane = threadIdx.x & 31;
    const float* hp = h + (size_t)n * HID + hd * HDIM;
    float s1 = hp[lane] * att_s[hd * HDIM + lane] + hp[lane + 32] * att_s[hd * HDIM + lane + 32];
    float s2 = hp[lane] * att_d[hd * HDIM + lane] + hp[lane + 32] * att_d[hd * HDIM + lane + 32];
#pragma unroll
    for (int o = 16; o > 0; o >>= 1) {
        s1 += __shfl_xor(s1, o, 32);
        s2 += __shfl_xor(s2, o, 32);
    }
    if (lane == 0) {
        asrc[(size_t)n * HEADS + hd] = s1;
        adst[(size_t)n * HEADS + hd] = s2;
    }
}

// ---------------------------------------------------------- edge: segment max
__global__ __launch_bounds__(256) void gat_edge_max(const int* __restrict__ src,
                                                    const int* __restrict__ dst,
                                                    const float* __restrict__ asrc,
                                                    const float* __restrict__ adst,
                                                    unsigned* __restrict__ menc,
                                                    int E, int N) {
    int e = blockIdx.x * 256 + threadIdx.x;
    if (e >= E + N) return;
    int s = (e < E) ? src[e] : (e - E);
    int d = (e < E) ? dst[e] : (e - E);
#pragma unroll
    for (int hd = 0; hd < HEADS; ++hd) {
        float l = asrc[(size_t)s * HEADS + hd] + adst[(size_t)d * HEADS + hd];
        l = (l < 0.0f) ? 0.2f * l : l;
        atomicMax(&menc[(size_t)d * HEADS + hd], fenc(l));
    }
}

// ------------------------------------------------- edge: exp and segment sum
__global__ __launch_bounds__(256) void gat_edge_expsum(const int* __restrict__ src,
                                                       const int* __restrict__ dst,
                                                       const float* __restrict__ asrc,
                                                       const float* __restrict__ adst,
                                                       const unsigned* __restrict__ menc,
                                                       float* __restrict__ ebuf,
                                                       float* __restrict__ denom,
                                                       int E, int N) {
    int e = blockIdx.x * 256 + threadIdx.x;
    if (e >= E + N) return;
    int s = (e < E) ? src[e] : (e - E);
    int d = (e < E) ? dst[e] : (e - E);
#pragma unroll
    for (int hd = 0; hd < HEADS; ++hd) {
        float l = asrc[(size_t)s * HEADS + hd] + adst[(size_t)d * HEADS + hd];
        l = (l < 0.0f) ? 0.2f * l : l;
        float m = fdec(menc[(size_t)d * HEADS + hd]);
        float w = __expf(l - m);
        ebuf[(size_t)e * HEADS + hd] = w;
        atomicAdd(&denom[(size_t)d * HEADS + hd], w);
    }
}

// --------------------------------------------- edge: weighted scatter-gather
// one block (256 threads) per edge; thread t handles channel t (head = t/64)
__global__ __launch_bounds__(256) void gat_edge_agg(const int* __restrict__ src,
                                                    const int* __restrict__ dst,
                                                    const float* __restrict__ ebuf,
                                                    const float* __restrict__ denom,
                                                    const float* __restrict__ h,
                                                    float* __restrict__ o,
                                                    int E, int N) {
    int e = blockIdx.x;
    int t = threadIdx.x;
    int s = (e < E) ? src[e] : (e - E);
    int d = (e < E) ? dst[e] : (e - E);
    int hd = t >> 6;
    float alpha = ebuf[(size_t)e * HEADS + hd] / denom[(size_t)d * HEADS + hd];
    atomicAdd(&o[(size_t)d * HID + t], alpha * h[(size_t)s * HID + t]);
}

// ------------------------------------------------- bias add + bf16 downconvert
__global__ __launch_bounds__(256) void gat_bias_cvt(const float* __restrict__ in,
                                                    const float* __restrict__ bias,
                                                    __bf16* __restrict__ out,
                                                    int N) {
    int n = blockIdx.x;
    int t = threadIdx.x;
    if (n >= N) return;
    size_t i = (size_t)n * HID + t;
    out[i] = (__bf16)(in[i] + bias[t]);
}

// ============================================================================
extern "C" void kernel_launch(void* const* d_in, const int* in_sizes, int n_in,
                              void* d_out, int out_size, void* d_ws, size_t ws_size,
                              hipStream_t stream) {
    (void)n_in; (void)out_size; (void)ws_size;
    const float* x   = (const float*)d_in[0];
    const int*   ei  = (const int*)d_in[1];      // [2,E]: src then dst
    const float* W1  = (const float*)d_in[3];
    const float* b1  = (const float*)d_in[4];
    const float* W2  = (const float*)d_in[5];
    const float* b2  = (const float*)d_in[6];
    const float* Wg1 = (const float*)d_in[7];
    const float* as1 = (const float*)d_in[8];
    const float* ad1 = (const float*)d_in[9];
    const float* bg1 = (const float*)d_in[10];
    const float* Wg2 = (const float*)d_in[11];
    const float* as2 = (const float*)d_in[12];
    const float* ad2 = (const float*)d_in[13];
    const float* bg2 = (const float*)d_in[14];

    const int N    = in_sizes[0] / IN_CH;
    const int E    = in_sizes[1] / 2;
    const int Etot = E + N;
    const int Mp   = (N + 31) & ~31;             // padded row count for WMMA inputs
    const int* srcI = ei;
    const int* dstI = ei + E;

    // ---- workspace carve-up (256B aligned); WMMA input buffers use Mp rows ----
    char* ws = (char*)d_ws;
    size_t off = 0;
    auto carve = [&](size_t bytes) -> char* {
        char* p = ws + off;
        off += (bytes + 255) & ~(size_t)255;
        return p;
    };
    // R0: xb (bf16, Mp*768) reused later as tf (f32, N*256) -- xb is larger
    char*   r0    = carve((size_t)Mp * IN_CH * 2);
    __bf16* xb    = (__bf16*)r0;
    float*  tf    = (float*)r0;                        // h @ Wg (per layer)
    __bf16* hb    = (__bf16*)carve((size_t)Mp * HID * 2);  // bf16 node features
    float*  of    = (float*)carve((size_t)N * HID * 4);    // GAT layer output f32
    float*  ebuf  = (float*)carve((size_t)Etot * HEADS * 4);
    float*  asrc  = (float*)carve((size_t)N * HEADS * 4);
    float*  adst  = (float*)carve((size_t)N * HEADS * 4);
    unsigned* menc = (unsigned*)carve((size_t)N * HEADS * 4);
    float*  denom = (float*)carve((size_t)N * HEADS * 4);
    __bf16* wbuf  = (__bf16*)carve((size_t)IN_CH * HID * 2); // transposed weights

    auto cdiv = [](long a, long b) { return (int)((a + b - 1) / b); };
    const int ggrid = cdiv(N, 32);

    // ---- linear1: h = leaky_relu(x @ W1 + b1, 0.01) -> hb (bf16) ----
    gat_cvt_bf16<<<cdiv((long)N * IN_CH, 256), 256, 0, stream>>>(x, xb, (long)N * IN_CH);
    gat_cvt_bf16_t<<<cdiv((long)IN_CH * HID, 256), 256, 0, stream>>>(W1, wbuf, IN_CH);
    gat_gemm_bf16<<<ggrid, 256, 0, stream>>>(xb, wbuf, b1, nullptr, hb,
                                             N, IN_CH, 0.01f);

    // ---- two GAT layers ----
    const float* Wg[2]  = {Wg1, Wg2};
    const float* asv[2] = {as1, as2};
    const float* adv[2] = {ad1, ad2};
    const float* bgv[2] = {bg1, bg2};
    for (int L = 0; L < 2; ++L) {
        gat_cvt_bf16_t<<<cdiv((long)HID * HID, 256), 256, 0, stream>>>(Wg[L], wbuf, HID);
        // tf = hb @ Wg  (f32, no bias/act)  -- xb is dead by now, tf aliases R0
        gat_gemm_bf16<<<ggrid, 256, 0, stream>>>(hb, wbuf, nullptr, tf, nullptr,
                                                 N, HID, 1.0f);
        gat_att_scores<<<N, 128, 0, stream>>>(tf, asv[L], adv[L], asrc, adst, N);

        hipMemsetAsync(menc, 0, (size_t)N * HEADS * 4, stream);
        hipMemsetAsync(denom, 0, (size_t)N * HEADS * 4, stream);
        hipMemsetAsync(of, 0, (size_t)N * HID * 4, stream);

        gat_edge_max<<<cdiv(Etot, 256), 256, 0, stream>>>(srcI, dstI, asrc, adst,
                                                          menc, E, N);
        gat_edge_expsum<<<cdiv(Etot, 256), 256, 0, stream>>>(srcI, dstI, asrc, adst,
                                                             menc, ebuf, denom, E, N);
        gat_edge_agg<<<Etot, 256, 0, stream>>>(srcI, dstI, ebuf, denom, tf, of, E, N);
        // of += bias_g ; hb = bf16(of)   (hb safe to overwrite: consumed above)
        gat_bias_cvt<<<N, 256, 0, stream>>>(of, bgv[L], hb, N);
    }

    // ---- linear2: out = leaky_relu(h @ W2 + b2, 0.01) -> d_out (f32) ----
    gat_cvt_bf16_t<<<cdiv((long)HID * HID, 256), 256, 0, stream>>>(W2, wbuf, HID);
    gat_gemm_bf16<<<ggrid, 256, 0, stream>>>(hb, wbuf, b2, (float*)d_out, nullptr,
                                             N, HID, 0.01f);
}
